// Transformer_12876311953620
// MI455X (gfx1250) — compile-verified, matrix operand order checked by reading, not run on hardware
//
#include <hip/hip_runtime.h>
#include <hip/hip_bf16.h>
#include <math.h>

// ---------------------------------------------------------------------------
// Types for CDNA5 WMMA (gfx1250, wave32)
// ---------------------------------------------------------------------------
typedef __bf16 bf16_t;
typedef __attribute__((ext_vector_type(16))) __bf16 v16bf;
typedef __attribute__((ext_vector_type(8)))  __bf16 v8bf;
typedef __attribute__((ext_vector_type(8)))  float   v8f;

#define ACT_NONE 0
#define ACT_SIG  1
#define ACT_TANH 2

static constexpr int BSZ  = 32;
static constexpr int SLEN = 128;
static constexpr int TLEN = 64;
static constexpr int DH   = 512;
static constexpr int D2   = 1024;
static constexpr int VOC  = 32000;
static constexpr int TSTEPS = TLEN - 1;                       // 63
static constexpr long long OUT_LD = (long long)TSTEPS * VOC;  // row stride of d_out per batch

// ---------------------------------------------------------------------------
// WMMA fragment loader: 16x32 bf16 tile (A or pre-transposed-B) from row-major
// [rows, ld] bf16 matrix.  Per the CDNA5 ISA 16-bit A layout:
//   lane L (0..15):  row = row0+L, elems 0..7 -> K k0+0..7,   elems 8..15 -> K k0+16..23
//   lane L (16..31): row = row0+L-16, elems 0..7 -> K k0+8..15, elems 8..15 -> K k0+24..31
// ---------------------------------------------------------------------------
__device__ __forceinline__ v16bf load_frag16x32(const bf16_t* __restrict__ base,
                                                int ld, int row0, int k0) {
  const int lane = threadIdx.x & 31;
  const int r  = row0 + (lane & 15);
  const int kk = k0 + ((lane & 16) >> 1);   // +8 for upper half-wave
  const bf16_t* p = base + (long long)r * ld + kk;
  union { v16bf v; v8bf h[2]; } u;
  u.h[0] = *(const v8bf*)(p);
  u.h[1] = *(const v8bf*)(p + 16);
  return u.v;
}

// ---------------------------------------------------------------------------
// Generic bf16 WMMA GEMM:  C[M,N] = act( initC + A[M,K] @ Bt[N,K]^T + bias )
//  - Bt is the weight stored TRANSPOSED ([N,K] row-major) so its fragment
//    loads identically to A.
//  - initC (fp32, stride ldi, may be negative, may alias C) seeds the
//    accumulator; nullptr -> zeros.
//  - One wave per 16x16 output tile; K-loop in steps of 32.
// ---------------------------------------------------------------------------
__global__ void k_gemm_bf16(const bf16_t* __restrict__ A, int lda,
                            const bf16_t* __restrict__ Bt, int ldb,
                            const float* __restrict__ bias,
                            const float* __restrict__ initC, long long ldi,
                            float* __restrict__ C, long long ldc,
                            int M, int N, int K, int act) {
  const int wave = (int)((blockIdx.x * blockDim.x + threadIdx.x) >> 5);
  const int lane = threadIdx.x & 31;
  const int tiles_m = M >> 4;
  const int tiles_n = N >> 4;
  if (wave >= tiles_m * tiles_n) return;      // wave-uniform: EXEC stays all-1s
  const int tm = wave % tiles_m;
  const int tn = wave / tiles_m;
  const int m0 = tm << 4, n0 = tn << 4;

  const int n     = n0 + (lane & 15);
  const int mbase = m0 + ((lane & 16) >> 1);  // +8 for upper half-wave (C layout)

  v8f c;
  if (initC) {
#pragma unroll
    for (int e = 0; e < 8; ++e) c[e] = initC[(long long)(mbase + e) * ldi + n];
  } else {
#pragma unroll
    for (int e = 0; e < 8; ++e) c[e] = 0.0f;
  }

  for (int k0 = 0; k0 < K; k0 += 32) {
    v16bf a = load_frag16x32(A,  lda, m0, k0);
    v16bf b = load_frag16x32(Bt, ldb, n0, k0);
    c = __builtin_amdgcn_wmma_f32_16x16x32_bf16(false, a, false, b, (short)0, c,
                                                false, false);
  }

  const float bn = bias ? bias[n] : 0.0f;
#pragma unroll
  for (int e = 0; e < 8; ++e) {
    float v = c[e] + bn;
    if (act == ACT_SIG)       v = 1.0f / (1.0f + __expf(-v));
    else if (act == ACT_TANH) v = tanhf(v);
    C[(long long)(mbase + e) * ldc + n] = v;
  }
}

// ---------------------------------------------------------------------------
// Small helper kernels
// ---------------------------------------------------------------------------
__global__ void k_transpose_bf16(const float* __restrict__ W, bf16_t* __restrict__ Wt,
                                 int K, int N) {   // W:[K,N] -> Wt:[N,K]
  long long i = (long long)blockIdx.x * blockDim.x + threadIdx.x;
  if (i >= (long long)K * N) return;
  int k = (int)(i / N), n = (int)(i % N);
  Wt[(long long)n * K + k] = (bf16_t)W[i];
}

__global__ void k_f32_to_bf16(const float* __restrict__ x, bf16_t* __restrict__ y, int n) {
  int i = blockIdx.x * blockDim.x + threadIdx.x;
  if (i < n) y[i] = (bf16_t)x[i];
}

__global__ void k_zero_f32(float* p, int n)  { int i = blockIdx.x*blockDim.x+threadIdx.x; if (i<n) p[i]=0.0f; }
__global__ void k_zero_bf16(bf16_t* p, int n){ int i = blockIdx.x*blockDim.x+threadIdx.x; if (i<n) p[i]=(bf16_t)0.0f; }
__global__ void k_zero_i32(int* p, int n)    { int i = blockIdx.x*blockDim.x+threadIdx.x; if (i<n) p[i]=0; }

// xs_bf[(s*B+b)*D + d] = bf16(E_src[x[b*S+s]*D + d])   (time-major like the scan)
__global__ void k_embed_src(const int* __restrict__ x, const float* __restrict__ E,
                            bf16_t* __restrict__ xs) {
  long long i = (long long)blockIdx.x * blockDim.x + threadIdx.x;
  if (i >= (long long)SLEN * BSZ * DH) return;
  int s   = (int)(i / (BSZ * DH));
  int rem = (int)(i % (BSZ * DH));
  int b   = rem / DH;
  int d   = rem % DH;
  xs[i] = (bf16_t)E[(long long)x[b * SLEN + s] * DH + d];
}

__global__ void k_embed_tgt(const int* __restrict__ y, const float* __restrict__ E,
                            bf16_t* __restrict__ w) {
  int i = blockIdx.x * blockDim.x + threadIdx.x;
  if (i >= BSZ * DH) return;
  int b = i / DH, d = i % DH;
  w[i] = (bf16_t)E[(long long)y[b] * DH + d];
}

__global__ void k_mul_to_bf16(const float* __restrict__ a, const float* __restrict__ b,
                              bf16_t* __restrict__ o, int n) {
  int i = blockIdx.x * blockDim.x + threadIdx.x;
  if (i < n) o[i] = (bf16_t)(a[i] * b[i]);
}

// h = (1-z)*h + z*hh ; also write bf16 copy and scatter into hmat slot
__global__ void k_gru_combine(const float* __restrict__ z, const float* __restrict__ hh,
                              float* __restrict__ h, bf16_t* __restrict__ h_bf,
                              float* __restrict__ hmat_slot, int hmat_stride) {
  int i = blockIdx.x * blockDim.x + threadIdx.x;
  if (i >= BSZ * DH) return;
  int b = i / DH, d = i % DH;
  float zn = z[i];
  float hn = (1.0f - zn) * h[i] + zn * hh[i];
  h[i] = hn;
  h_bf[i] = (bf16_t)hn;
  hmat_slot[(long long)b * hmat_stride + d] = hn;
}

// s = (1-z)*s + z*st ; also bf16 copy
__global__ void k_dec_combine(const float* __restrict__ z, const float* __restrict__ st,
                              float* __restrict__ s, bf16_t* __restrict__ s_bf, int n) {
  int i = blockIdx.x * blockDim.x + threadIdx.x;
  if (i >= n) return;
  float zn = z[i];
  float sn = (1.0f - zn) * s[i] + zn * st[i];
  s[i] = sn;
  s_bf[i] = (bf16_t)sn;
}

// e[b,s] = att_v.W . tanh(ss[b,:] + hproj[b*S+s,:]) + att_v.b   (wave per (b,s))
__global__ void k_att_e(const float* __restrict__ ss, const float* __restrict__ hproj,
                        const float* __restrict__ vW, const float* __restrict__ vb,
                        float* __restrict__ e) {
  int wave = (int)((blockIdx.x * blockDim.x + threadIdx.x) >> 5);
  int lane = threadIdx.x & 31;
  if (wave >= BSZ * SLEN) return;
  int b = wave / SLEN;
  const float* hp = hproj + (long long)wave * DH;
  const float* sb = ss + (long long)b * DH;
  float acc = 0.0f;
  for (int d = lane; d < DH; d += 32) acc += vW[d] * tanhf(sb[d] + hp[d]);
  for (int off = 16; off; off >>= 1) acc += __shfl_down(acc, off, 32);
  if (lane == 0) e[wave] = acc + vb[0];
}

// softmax over S=128 per batch row (one 128-thread block per b)
__global__ void k_softmax128(const float* __restrict__ e, float* __restrict__ a) {
  __shared__ float buf[128];
  int b = blockIdx.x, t = threadIdx.x;
  float v = e[b * SLEN + t];
  buf[t] = v; __syncthreads();
  for (int off = 64; off; off >>= 1) { if (t < off) buf[t] = fmaxf(buf[t], buf[t + off]); __syncthreads(); }
  float mx = buf[0]; __syncthreads();
  float ex = __expf(v - mx);
  buf[t] = ex; __syncthreads();
  for (int off = 64; off; off >>= 1) { if (t < off) buf[t] += buf[t + off]; __syncthreads(); }
  a[b * SLEN + t] = ex / buf[0];
}

// c[b,d] = sum_s a[b,s] * hmat[(b*S+s)*D2 + d] ; also bf16 copy
__global__ void k_context(const float* __restrict__ a, const float* __restrict__ hmat,
                          float* __restrict__ c, bf16_t* __restrict__ c_bf) {
  int i = blockIdx.x * blockDim.x + threadIdx.x;
  if (i >= BSZ * D2) return;
  int b = i >> 10, d = i & (D2 - 1);
  const float* ab = a + b * SLEN;
  const float* hb = hmat + (long long)b * SLEN * D2 + d;
  float acc = 0.0f;
  for (int s = 0; s < SLEN; ++s) acc += ab[s] * hb[(long long)s * D2];
  c[i] = acc;
  c_bf[i] = (bf16_t)acc;
}

// t[b,j] = max(tt[b,2j], tt[b,2j+1]) ; also bf16 copy
__global__ void k_maxpool2(const float* __restrict__ tt, float* __restrict__ t,
                           bf16_t* __restrict__ t_bf) {
  int i = blockIdx.x * blockDim.x + threadIdx.x;
  if (i >= BSZ * DH) return;
  int b = i >> 9, j = i & (DH - 1);
  float v = fmaxf(tt[b * D2 + 2 * j], tt[b * D2 + 2 * j + 1]);
  t[i] = v;
  t_bf[i] = (bf16_t)v;
}

// first-occurrence argmax over V per batch row (one 256-thread block per b)
__global__ void k_argmax(const float* __restrict__ out, long long ldc, int* __restrict__ y) {
  __shared__ float sv[256];
  __shared__ int   si[256];
  int b = blockIdx.x, t = threadIdx.x;
  const float* row = out + (long long)b * ldc;
  float best = -INFINITY; int bi = 0x7fffffff;
  for (int v = t; v < VOC; v += 256) {
    float xv = row[v];
    if (xv > best) { best = xv; bi = v; }
  }
  sv[t] = best; si[t] = bi; __syncthreads();
  for (int off = 128; off; off >>= 1) {
    if (t < off) {
      if (sv[t + off] > sv[t] || (sv[t + off] == sv[t] && si[t + off] < si[t])) {
        sv[t] = sv[t + off]; si[t] = si[t + off];
      }
    }
    __syncthreads();
  }
  if (t == 0) y[b] = si[0];
}

// ---------------------------------------------------------------------------
// Host-side orchestration
// ---------------------------------------------------------------------------
extern "C" void kernel_launch(void* const* d_in, const int* in_sizes, int n_in,
                              void* d_out, int out_size, void* d_ws, size_t ws_size,
                              hipStream_t stream) {
  (void)in_sizes; (void)n_in; (void)out_size; (void)ws_size;

  const int*   x_tok = (const int*)d_in[0];
  const float* E_src = (const float*)d_in[3];
  const float* E_tgt = (const float*)d_in[4];
  auto P = [&](int i) { return (const float*)d_in[i]; };
  float* out = (float*)d_out;

  // ---- workspace bump allocator (256B aligned) ----
  char* base = (char*)d_ws;
  size_t off = 0;
  auto alloc = [&](size_t bytes) -> void* {
    void* p = base + off;
    off = (off + bytes + 255) & ~(size_t)255;
    return p;
  };

  // transposed bf16 weights
  bf16_t* wt_gru[12];
  for (int i = 0; i < 12; ++i) wt_gru[i] = (bf16_t*)alloc((size_t)DH * DH * 2);
  bf16_t* wt_att_s = (bf16_t*)alloc((size_t)DH * DH * 2);
  bf16_t* wt_att_h = (bf16_t*)alloc((size_t)DH * D2 * 2);   // [512,1024]
  bf16_t* wt_t_s   = (bf16_t*)alloc((size_t)D2 * DH * 2);   // [1024,512]
  bf16_t* wt_t_y   = (bf16_t*)alloc((size_t)D2 * DH * 2);
  bf16_t* wt_t_c   = (bf16_t*)alloc((size_t)D2 * D2 * 2);
  bf16_t* wt_out   = (bf16_t*)alloc((size_t)VOC * DH * 2);  // [32000,512]
  bf16_t* wt_z_y   = (bf16_t*)alloc((size_t)DH * DH * 2);
  bf16_t* wt_z_s   = (bf16_t*)alloc((size_t)DH * DH * 2);
  bf16_t* wt_z_c   = (bf16_t*)alloc((size_t)DH * D2 * 2);   // [512,1024]
  bf16_t* wt_r_y   = (bf16_t*)alloc((size_t)DH * DH * 2);
  bf16_t* wt_r_s   = (bf16_t*)alloc((size_t)DH * DH * 2);
  bf16_t* wt_st_y  = (bf16_t*)alloc((size_t)DH * DH * 2);
  bf16_t* wt_st_s  = (bf16_t*)alloc((size_t)DH * DH * 2);
  bf16_t* wt_st_c  = (bf16_t*)alloc((size_t)DH * D2 * 2);

  // activations
  const int SB = SLEN * BSZ;                                 // 4096
  bf16_t* xs_bf   = (bf16_t*)alloc((size_t)SB * DH * 2);
  float*  xproj[6];                                          // r_z,r_r,r_h,l_z,l_r,l_h
  for (int i = 0; i < 6; ++i) xproj[i] = (float*)alloc((size_t)SB * DH * 4);
  float*  hmat    = (float*) alloc((size_t)SB * D2 * 4);
  bf16_t* hmat_bf = (bf16_t*)alloc((size_t)SB * D2 * 2);
  float*  hproj   = (float*) alloc((size_t)SB * DH * 4);
  float*  h_st[2]; bf16_t* h_bf[2];
  for (int i = 0; i < 2; ++i) {
    h_st[i] = (float*) alloc((size_t)BSZ * DH * 4);
    h_bf[i] = (bf16_t*)alloc((size_t)BSZ * DH * 2);
  }
  float*  zbuf  = (float*) alloc((size_t)BSZ * DH * 4);
  float*  rbuf  = (float*) alloc((size_t)BSZ * DH * 4);
  float*  hhat  = (float*) alloc((size_t)BSZ * DH * 4);
  bf16_t* rh_bf = (bf16_t*)alloc((size_t)BSZ * DH * 2);
  float*  svec  = (float*) alloc((size_t)BSZ * DH * 4);
  bf16_t* s_bf  = (bf16_t*)alloc((size_t)BSZ * DH * 2);
  float*  ssv   = (float*) alloc((size_t)BSZ * DH * 4);
  float*  evec  = (float*) alloc((size_t)BSZ * SLEN * 4);
  float*  avec  = (float*) alloc((size_t)BSZ * SLEN * 4);
  float*  cvec  = (float*) alloc((size_t)BSZ * D2 * 4);
  bf16_t* c_bf  = (bf16_t*)alloc((size_t)BSZ * D2 * 2);
  bf16_t* w_bf  = (bf16_t*)alloc((size_t)BSZ * DH * 2);
  float*  ttil  = (float*) alloc((size_t)BSZ * D2 * 4);
  float*  tvec  = (float*) alloc((size_t)BSZ * DH * 4);
  bf16_t* t_bf  = (bf16_t*)alloc((size_t)BSZ * DH * 2);
  float*  zg    = (float*) alloc((size_t)BSZ * DH * 4);
  float*  rg    = (float*) alloc((size_t)BSZ * DH * 4);
  float*  stg   = (float*) alloc((size_t)BSZ * DH * 4);
  bf16_t* rs_bf = (bf16_t*)alloc((size_t)BSZ * DH * 2);
  int*    ybuf  = (int*)   alloc((size_t)BSZ * 4);

  auto blocks = [](long long n) { return (unsigned)((n + 255) / 256); };

  auto gemm = [&](const bf16_t* A, int lda, const bf16_t* Bt, int ldb,
                  const float* bias, const float* initC, long long ldi,
                  float* C, long long ldc, int M, int N, int K, int act) {
    int tiles = (M >> 4) * (N >> 4);
    k_gemm_bf16<<<(tiles + 7) / 8, 256, 0, stream>>>(A, lda, Bt, ldb, bias,
                                                     initC, ldi, C, ldc, M, N, K, act);
  };
  auto tr = [&](const float* Wp, int K, int N, bf16_t* Wt) {
    k_transpose_bf16<<<blocks((long long)K * N), 256, 0, stream>>>(Wp, Wt, K, N);
  };

  // ---- 1) convert all weights once ----
  const float* gruW[12] = { P(5),P(7),P(9),P(11),P(13),P(15), P(17),P(19),P(21),P(23),P(25),P(27) };
  const float* gruB[12] = { P(6),P(8),P(10),P(12),P(14),P(16), P(18),P(20),P(22),P(24),P(26),P(28) };
  for (int i = 0; i < 12; ++i) tr(gruW[i], DH, DH, wt_gru[i]);
  tr(P(29), DH, DH, wt_att_s);
  tr(P(31), D2, DH, wt_att_h);
  tr(P(35), DH, D2, wt_t_s);
  tr(P(37), DH, D2, wt_t_y);
  tr(P(39), D2, D2, wt_t_c);
  tr(P(41), DH, VOC, wt_out);
  tr(P(43), DH, DH, wt_z_y);
  tr(P(45), DH, DH, wt_z_s);
  tr(P(47), D2, DH, wt_z_c);
  tr(P(49), DH, DH, wt_r_y);
  tr(P(51), DH, DH, wt_r_s);            // r_c (idx 53) unused: reference bug reuses z_c
  tr(P(55), DH, DH, wt_st_y);
  tr(P(57), DH, DH, wt_st_s);
  tr(P(59), D2, DH, wt_st_c);

  // ---- 2) source embedding, time-major bf16 ----
  k_embed_src<<<blocks((long long)SB * DH), 256, 0, stream>>>(x_tok, E_src, xs_bf);

  // ---- 3) precompute x-side GRU projections (incl. x-side biases) ----
  // xproj[side*3 + g] = xs @ W_{g}x + b_{g}x ; g order z,r,h via gru idx {0,2,4}
  for (int side = 0; side < 2; ++side)
    for (int g = 0; g < 3; ++g) {
      int wi = side * 6 + g * 2;        // zx=0, rx=2, hx=4 (+6 for left)
      gemm(xs_bf, DH, wt_gru[wi], DH, gruB[wi], nullptr, 0,
           xproj[side * 3 + g], DH, SB, DH, DH, ACT_NONE);
    }

  // ---- 4) bidirectional GRU scans ----
  auto scan = [&](int side, int hmat_off, bool flip) {
    float*  h   = h_st[side];
    bf16_t* hbf = h_bf[side];
    k_zero_f32 <<<blocks(BSZ * DH), 256, 0, stream>>>(h,   BSZ * DH);
    k_zero_bf16<<<blocks(BSZ * DH), 256, 0, stream>>>(hbf, BSZ * DH);
    int wzh = side * 6 + 1, wrh = side * 6 + 3, whh = side * 6 + 5;
    for (int s = 0; s < SLEN; ++s) {
      long long ldi = flip ? -(long long)DH : (long long)DH;
      long long ro  = flip ? ((long long)s * BSZ + (BSZ - 1)) * DH
                           :  (long long)s * BSZ * DH;
      const float* iz = xproj[side * 3 + 0] + ro;
      const float* ir = xproj[side * 3 + 1] + ro;
      const float* ih = xproj[side * 3 + 2] + ro;
      gemm(hbf, DH, wt_gru[wzh], DH, gruB[wzh], iz, ldi, zbuf, DH, BSZ, DH, DH, ACT_SIG);
      gemm(hbf, DH, wt_gru[wrh], DH, gruB[wrh], ir, ldi, rbuf, DH, BSZ, DH, DH, ACT_SIG);
      k_mul_to_bf16<<<blocks(BSZ * DH), 256, 0, stream>>>(rbuf, h, rh_bf, BSZ * DH);
      gemm(rh_bf, DH, wt_gru[whh], DH, gruB[whh], ih, ldi, hhat, DH, BSZ, DH, DH, ACT_TANH);
      k_gru_combine<<<blocks(BSZ * DH), 256, 0, stream>>>(
          zbuf, hhat, h, hbf, hmat + (long long)s * D2 + hmat_off, SLEN * D2);
    }
  };
  scan(0, 0,  false);   // 'right' cell -> h_left  -> hmat[..., 0:512]
  scan(1, DH, true);    // 'left' cell, batch-flipped input -> hmat[..., 512:1024]

  // ---- 5) hproj = hmat @ att_h + b (step-invariant) ----
  k_f32_to_bf16<<<blocks((long long)SB * D2), 256, 0, stream>>>(hmat, hmat_bf, SB * D2);
  gemm(hmat_bf, D2, wt_att_h, D2, P(32), nullptr, 0, hproj, DH, SB, DH, D2, ACT_NONE);

  // ---- 6) decoder ----
  k_zero_f32 <<<blocks(BSZ * DH), 256, 0, stream>>>(svec, BSZ * DH);
  k_zero_bf16<<<blocks(BSZ * DH), 256, 0, stream>>>(s_bf, BSZ * DH);
  k_zero_i32 <<<1, 32, 0, stream>>>(ybuf, BSZ);

  for (int step = 0; step < TSTEPS; ++step) {
    // attention
    gemm(s_bf, DH, wt_att_s, DH, P(30), nullptr, 0, ssv, DH, BSZ, DH, DH, ACT_NONE);
    k_att_e<<<(BSZ * SLEN) / 8, 256, 0, stream>>>(ssv, hproj, P(33), P(34), evec);
    k_softmax128<<<BSZ, 128, 0, stream>>>(evec, avec);
    k_context<<<blocks(BSZ * D2), 256, 0, stream>>>(avec, hmat, cvec, c_bf);

    // target embedding of previous token
    k_embed_tgt<<<blocks(BSZ * DH), 256, 0, stream>>>(ybuf, E_tgt, w_bf);

    // t_tilde = s@t_s + w@t_y + c@t_c (+ all biases); maxpool(2)
    gemm(s_bf, DH, wt_t_s, DH, P(36), nullptr, 0,    ttil, D2, BSZ, D2, DH, ACT_NONE);
    gemm(w_bf, DH, wt_t_y, DH, P(38), ttil, D2,      ttil, D2, BSZ, D2, DH, ACT_NONE);
    gemm(c_bf, D2, wt_t_c, D2, P(40), ttil, D2,      ttil, D2, BSZ, D2, D2, ACT_NONE);
    k_maxpool2<<<blocks(BSZ * DH), 256, 0, stream>>>(ttil, tvec, t_bf);

    // logits for this step -> d_out[:, step, :]
    gemm(t_bf, DH, wt_out, DH, P(42), nullptr, 0,
         out + (long long)step * VOC, OUT_LD, BSZ, VOC, DH, ACT_NONE);

    // DeRNN gates (r reuses z_c weights+bias, bug-faithful)
    gemm(w_bf, DH, wt_z_y, DH, P(44), nullptr, 0, zg, DH, BSZ, DH, DH, ACT_NONE);
    gemm(s_bf, DH, wt_z_s, DH, P(46), zg, DH,     zg, DH, BSZ, DH, DH, ACT_NONE);
    gemm(c_bf, D2, wt_z_c, D2, P(48), zg, DH,     zg, DH, BSZ, DH, D2, ACT_SIG);

    gemm(w_bf, DH, wt_r_y, DH, P(50), nullptr, 0, rg, DH, BSZ, DH, DH, ACT_NONE);
    gemm(s_bf, DH, wt_r_s, DH, P(52), rg, DH,     rg, DH, BSZ, DH, DH, ACT_NONE);
    gemm(c_bf, D2, wt_z_c, D2, P(48), rg, DH,     rg, DH, BSZ, DH, D2, ACT_SIG);

    k_mul_to_bf16<<<blocks(BSZ * DH), 256, 0, stream>>>(rg, svec, rs_bf, BSZ * DH);
    gemm(w_bf,  DH, wt_st_y, DH, P(56), nullptr, 0, stg, DH, BSZ, DH, DH, ACT_NONE);
    gemm(rs_bf, DH, wt_st_s, DH, P(58), stg, DH,    stg, DH, BSZ, DH, DH, ACT_NONE);
    gemm(c_bf,  D2, wt_st_c, D2, P(60), stg, DH,    stg, DH, BSZ, DH, D2, ACT_TANH);

    // greedy feedback + state update
    k_argmax<<<BSZ, 256, 0, stream>>>(out + (long long)step * VOC, OUT_LD, ybuf);
    k_dec_combine<<<blocks(BSZ * DH), 256, 0, stream>>>(zg, stg, svec, s_bf, BSZ * DH);
  }
}
